// AGCB_Patch_my1_84189948936443
// MI455X (gfx1250) — matrix operand order repaired and checked by reading, not compiled, not run on hardware
//
#include <hip/hip_runtime.h>
#include <hip/hip_bf16.h>

// ---------------------------------------------------------------------------
// AGCB_Patch pipeline for MI455X (gfx1250, wave32, WMMA).
// All matmuls (1x1 convs, criss-cross energies, attention-apply, 3x3 conv)
// run on v_wmma_f32_16x16x32_bf16 with f32 accumulation; operand tiles are
// staged in LDS. Where the global tile is contiguous we use the CDNA5 async
// global->LDS path (GLOBAL_LOAD_ASYNC_TO_LDS_*, ASYNCcnt) when the toolchain
// exposes the builtins; otherwise we fall back to synchronous staging.
// ---------------------------------------------------------------------------

typedef __attribute__((ext_vector_type(16))) __bf16 v16bf;
typedef __attribute__((ext_vector_type(8)))  float  v8f;
typedef __attribute__((ext_vector_type(2)))  int    i32x2;
typedef __attribute__((ext_vector_type(4)))  int    i32x4;

union Frag { v16bf v; unsigned int u[8]; __bf16 h[16]; };

#define BDIM 8
#define CDIM 256
#define HDIM 128
#define WDIM 128
#define NBLK 32      // B * s * s
#define HS 64
#define WS 64
#define NP 4096      // HS*WS
#define CR 32
#define CN 8
#define NEGBIG (-1e30f)

// ---- CDNA5 async global->LDS support (guarded; fallback = sync staging) ---
#if defined(__has_builtin)
# if __has_builtin(__builtin_amdgcn_global_load_async_to_lds_b64) && \
     __has_builtin(__builtin_amdgcn_global_load_async_to_lds_b128)
#  define HAVE_ASYNC_LDS 1
# else
#  define HAVE_ASYNC_LDS 0
# endif
#else
# define HAVE_ASYNC_LDS 0
#endif

#define GAS __attribute__((address_space(1)))
#define LAS __attribute__((address_space(3)))

#if HAVE_ASYNC_LDS
__device__ __forceinline__ void async_copy_b64(const void* g, void* l) {
  __builtin_amdgcn_global_load_async_to_lds_b64((GAS i32x2*)g, (LAS i32x2*)l, 0, 0);
}
__device__ __forceinline__ void async_copy_b128(const void* g, void* l) {
  __builtin_amdgcn_global_load_async_to_lds_b128((GAS i32x4*)g, (LAS i32x4*)l, 0, 0);
}
__device__ __forceinline__ void wait_async0() {
# if __has_builtin(__builtin_amdgcn_s_wait_asynccnt)
  __builtin_amdgcn_s_wait_asynccnt(0);
# else
  asm volatile("s_wait_asynccnt 0x0" ::: "memory");
# endif
}
#endif

// ---- WMMA fragment builders from LDS tiles -------------------------------
// A (16x32 bf16): lane = half*16+ln holds row M=ln; VGPR j -> K=half*16+2j.
// Pairs are contiguous -> 32-bit LDS reads.
__device__ __forceinline__ v16bf load_frag_a(const __bf16* lds, int m, int half) {
  Frag f;
#pragma unroll
  for (int j = 0; j < 8; ++j) {
    int k = half * 16 + 2 * j;
    f.u[j] = *(const unsigned int*)&lds[m * 32 + k];
  }
  return f.v;
}
// B (32x16 bf16) stored N-major in LDS ([n][32]): lane holds column n=ln,
// K = half*16 + 2j (+0,+1) (contiguous k, same order as load_frag_a).
__device__ __forceinline__ v16bf load_frag_b(const __bf16* lds, int n, int half) {
  Frag f;
#pragma unroll
  for (int j = 0; j < 8; ++j) {
    int k = half * 16 + 2 * j;
    f.u[j] = *(const unsigned int*)&lds[n * 32 + k];
  }
  return f.v;
}
// B tile kept k-major ([k][64] bf16, as async-copied): 16-bit LDS reads.
__device__ __forceinline__ v16bf load_frag_b_km(const __bf16* lds, int n, int half) {
  Frag f;
#pragma unroll
  for (int j = 0; j < 8; ++j) {
    int k = half * 16 + 2 * j;
    f.h[2 * j]     = lds[k * 64 + n];
    f.h[2 * j + 1] = lds[(k + 1) * 64 + n];
  }
  return f.v;
}
// B tile kept k-major as raw f32 ([k][64], async-copied): convert at build.
__device__ __forceinline__ v16bf load_frag_b_f32km(const float* lds, int n, int half) {
  Frag f;
#pragma unroll
  for (int j = 0; j < 8; ++j) {
    int k = half * 16 + 2 * j;
    f.h[2 * j]     = (__bf16)lds[k * 64 + n];
    f.h[2 * j + 1] = (__bf16)lds[(k + 1) * 64 + n];
  }
  return f.v;
}

// ---- adaptive max pool to (B,C,2,2) --------------------------------------
__global__ void pool_max_k(const float* __restrict__ x, float* __restrict__ pooled) {
  __shared__ float red[256];
  int bid = blockIdx.x;                       // ((b*256+c)*2+si)*2+sj
  int b = bid >> 10, c = (bid >> 2) & 255, si = (bid >> 1) & 1, sj = bid & 1;
  const float* src = x + ((size_t)(b * CDIM + c) * HDIM + si * HS) * WDIM + sj * WS;
  float mx = NEGBIG;
#pragma unroll
  for (int i = 0; i < 16; ++i) {
    int e = i * 256 + threadIdx.x;
    mx = fmaxf(mx, src[(e >> 6) * WDIM + (e & 63)]);
  }
  red[threadIdx.x] = mx; __syncthreads();
  for (int o = 128; o > 0; o >>= 1) {
    if ((int)threadIdx.x < o) red[threadIdx.x] = fmaxf(red[threadIdx.x], red[threadIdx.x + o]);
    __syncthreads();
  }
  if (threadIdx.x == 0) pooled[bid] = red[0];
}

// ---- tiny nonlocal block + sigmoid -> channel gates (B,C,2,2) ------------
__global__ void gca_k(const float* __restrict__ pooled,
                      const float* __restrict__ nqw, const float* __restrict__ nqb,
                      const float* __restrict__ nkw, const float* __restrict__ nkb,
                      const float* __restrict__ nvw, const float* __restrict__ nvb,
                      const float* __restrict__ nlg, float* __restrict__ gca) {
  __shared__ float ps[CDIM][4];
  __shared__ float qs[CN][4], ks2[CN][4];
  __shared__ float att[4][4];
  int b = blockIdx.x, c = threadIdx.x;
#pragma unroll
  for (int m = 0; m < 4; ++m) ps[c][m] = pooled[(b * CDIM + c) * 4 + m];
  __syncthreads();
  if (c < CN) {
    float aq[4] = {nqb[c], nqb[c], nqb[c], nqb[c]};
    float ak[4] = {nkb[c], nkb[c], nkb[c], nkb[c]};
    for (int c2 = 0; c2 < CDIM; ++c2) {
      float wq_ = nqw[c * CDIM + c2], wk_ = nkw[c * CDIM + c2];
#pragma unroll
      for (int m = 0; m < 4; ++m) { aq[m] += wq_ * ps[c2][m]; ak[m] += wk_ * ps[c2][m]; }
    }
#pragma unroll
    for (int m = 0; m < 4; ++m) { qs[c][m] = aq[m]; ks2[c][m] = ak[m]; }
  }
  __syncthreads();
  if (c < 16) {
    int n = c >> 2, m = c & 3;
    float e = 0.f;
    for (int cc = 0; cc < CN; ++cc) e += qs[cc][n] * ks2[cc][m];
    att[n][m] = e;
  }
  __syncthreads();
  if (c < 4) {
    float mx = fmaxf(fmaxf(att[c][0], att[c][1]), fmaxf(att[c][2], att[c][3]));
    float s = 0.f, ex[4];
#pragma unroll
    for (int m = 0; m < 4; ++m) { ex[m] = __expf(att[c][m] - mx); s += ex[m]; }
#pragma unroll
    for (int m = 0; m < 4; ++m) att[c][m] = ex[m] / s;
  }
  __syncthreads();
  float vv[4] = {nvb[c], nvb[c], nvb[c], nvb[c]};
  for (int c2 = 0; c2 < CDIM; ++c2) {
    float wv_ = nvw[c * CDIM + c2];
#pragma unroll
    for (int m = 0; m < 4; ++m) vv[m] += wv_ * ps[c2][m];
  }
  float g = nlg[0];
#pragma unroll
  for (int n = 0; n < 4; ++n) {
    float o = 0.f;
#pragma unroll
    for (int m = 0; m < 4; ++m) o += vv[m] * att[n][m];
    float z = g * o + ps[c][n];
    gca[(b * CDIM + c) * 4 + n] = 1.f / (1.f + __expf(-z));
  }
}

// ---- fold s*s patches into batch dim: x(B,C,H,W) -> xb(32,C,64,64) -------
__global__ void extract_blocks_k(const float* __restrict__ x, float* __restrict__ xb) {
  int idx = blockIdx.x * 256 + threadIdx.x;        // 32*256*4096 = 2^25
  int bb = idx >> 20, r = idx & 0xFFFFF, c = r >> 12, p = r & 4095;
  int y = p >> 6, xx = p & 63;
  int b = bb >> 2, si = (bb >> 1) & 1, sj = bb & 1;
  xb[idx] = x[((size_t)(b * CDIM + c) * HDIM + si * HS + y) * WDIM + sj * WS + xx];
}

// ---- nearest rotation about the 64x64 patch center -----------------------
__global__ void rotate_k(const float* __restrict__ src, float* __restrict__ dst,
                         float ct, float st) {
  int idx = blockIdx.x * 256 + threadIdx.x;
  int base = idx >> 12, p = idx & 4095;
  int y = p >> 6, xx = p & 63;
  float fx = (float)xx - 31.5f, fy = (float)y - 31.5f;
  float sx =  ct * fx + st * fy + 31.5f;
  float sy = -st * fx + ct * fy + 31.5f;
  float rx = rintf(sx), ry = rintf(sy);           // round-half-even == jnp.round
  float v = 0.f;
  if (rx >= 0.f && rx < 64.f && ry >= 0.f && ry < 64.f)
    v = src[(base << 12) + (int)ry * WS + (int)rx];
  dst[idx] = v;
}

// ---- 1x1 conv as GEMM: Out[bb,M,NP](bf16) = W(MxC) * X[bb](CxNP) + bias ---
// 8 waves (2m x 4n) per block -> 32x64 tile, K looped 256 in steps of 32.
// B tile (f32, 8KB, k-major) arrives via async global->LDS when available.
__global__ void gemm1x1_k(const float* __restrict__ Wm, const float* __restrict__ bias,
                          const float* __restrict__ X, __bf16* __restrict__ Out, int M) {
  __shared__ __bf16 As[32 * 32];
#if HAVE_ASYNC_LDS
  __shared__ __align__(16) float Bf[32 * 64];
#else
  __shared__ __bf16 Bs[64 * 32];
#endif
  int bb = blockIdx.x, n0 = blockIdx.y * 64, m0 = blockIdx.z * 32;
  int t = threadIdx.x, lane = t & 31, wid = t >> 5;
  int wm = wid & 1, wn = wid >> 1;
  int half = lane >> 4, ln = lane & 15;
  v8f acc = {};
  for (int ks = 0; ks < 8; ++ks) {
    int c0 = ks * 32;
#if HAVE_ASYNC_LDS
    {  // row k is 64 contiguous f32 (256B) -> 2x b128 per lane, 256 lanes
      int k = t >> 3, chunk = t & 7;
      const char* g = (const char*)&X[(size_t)(bb * CDIM + c0 + k) * NP + n0] + chunk * 32;
      char* l = (char*)&Bf[k * 64] + chunk * 32;
      async_copy_b128(g, l);
      async_copy_b128(g + 16, l + 16);
    }
#endif
#pragma unroll
    for (int i = 0; i < 4; ++i) {
      int e = i * 256 + t, m = e >> 5, k = e & 31;
      As[m * 32 + k] = (__bf16)Wm[(m0 + m) * CDIM + c0 + k];
    }
#if !HAVE_ASYNC_LDS
#pragma unroll
    for (int i = 0; i < 8; ++i) {
      int e = i * 256 + t, n = e & 63, k = e >> 6;
      Bs[n * 32 + k] = (__bf16)X[(size_t)(bb * CDIM + c0 + k) * NP + n0 + n];
    }
#endif
    if (ks < 7)  // warm L2 for the next K-slab (global_prefetch_b8)
      __builtin_prefetch(&X[(size_t)(bb * CDIM + c0 + 32) * NP + n0 + lane], 0, 1);
#if HAVE_ASYNC_LDS
    wait_async0();
#endif
    __syncthreads();
    v16bf af = load_frag_a(As, wm * 16 + ln, half);
#if HAVE_ASYNC_LDS
    v16bf bf_ = load_frag_b_f32km(Bf, wn * 16 + ln, half);
#else
    v16bf bf_ = load_frag_b(Bs, wn * 16 + ln, half);
#endif
    acc = __builtin_amdgcn_wmma_f32_16x16x32_bf16(false, af, false, bf_, (short)0, acc, false, false);
    __syncthreads();
  }
  int n = n0 + wn * 16 + ln;
#pragma unroll
  for (int r2 = 0; r2 < 8; ++r2) {
    int m = m0 + wm * 16 + half * 8 + r2;
    Out[(size_t)(bb * M + m) * NP + n] = (__bf16)(acc[r2] + bias[m]);
  }
}

// ---- criss-cross energies: 64x64 = Q^T K over Cr=32, per (bb, fixed h/w) --
// fixedIsW=1: eH[bb,w,h,g]; fixedIsW=0: eW[bb,h,w,v]. Two fixed idx / block.
__global__ void energy_k(const __bf16* __restrict__ Q, const __bf16* __restrict__ K,
                         float* __restrict__ E, int fixedIsW) {
  __shared__ __bf16 Aq[2][64 * 32];
  __shared__ __bf16 Bk[2][64 * 32];
  int bb = blockIdx.x;
  int hfix = threadIdx.x >> 7;
  int fix = blockIdx.y * 2 + hfix;
  int t = threadIdx.x & 127;
  int wid4 = t >> 5, lane = t & 31, half = lane >> 4, ln = lane & 15;
#pragma unroll
  for (int i = 0; i < 16; ++i) {
    int e = i * 128 + t, m = e >> 5, k = e & 31;
    int pix = fixedIsW ? (m * WS + fix) : (fix * WS + m);
    Aq[hfix][m * 32 + k] = Q[(size_t)(bb * CR + k) * NP + pix];
    Bk[hfix][m * 32 + k] = K[(size_t)(bb * CR + k) * NP + pix];
  }
  __syncthreads();
  v16bf af = load_frag_a(Aq[hfix], wid4 * 16 + ln, half);
  float* eout = E + (size_t)((bb * 64 + fix) * 64) * 64;
#pragma unroll
  for (int gt = 0; gt < 4; ++gt) {
    v8f acc = {};
    v16bf bf_ = load_frag_b(Bk[hfix], gt * 16 + ln, half);
    acc = __builtin_amdgcn_wmma_f32_16x16x32_bf16(false, af, false, bf_, (short)0, acc, false, false);
    int n = gt * 16 + ln;
#pragma unroll
    for (int r2 = 0; r2 < 8; ++r2) {
      int m = wid4 * 16 + half * 8 + r2;
      eout[m * 64 + n] = acc[r2];
    }
  }
}

// ---- joint softmax over concat(eH[64] masked at g==h, eW[64]) ------------
// one wave per pixel; writes attH (bb,w,g,h) and attW (bb,h,v,w) as bf16.
__global__ void softmax_k(const float* __restrict__ EH, const float* __restrict__ EW,
                          __bf16* __restrict__ AH, __bf16* __restrict__ AW) {
  int wid = threadIdx.x >> 5, lane = threadIdx.x & 31;
  int p = blockIdx.x * 8 + wid;
  int bb = p >> 12, rem = p & 4095, h = rem >> 6, w = rem & 63;
  const float* eh = EH + (size_t)(((bb * 64 + w) * 64 + h)) * 64;
  const float* ew = EW + (size_t)(((bb * 64 + h) * 64 + w)) * 64;
  float vals[4];
  float mx = NEGBIG;
#pragma unroll
  for (int i = 0; i < 4; ++i) {
    int j = lane + i * 32;
    float v = (j < 64) ? ((j == h) ? NEGBIG : eh[j]) : ew[j - 64];
    vals[i] = v; mx = fmaxf(mx, v);
  }
#pragma unroll
  for (int o = 16; o > 0; o >>= 1) mx = fmaxf(mx, __shfl_xor(mx, o, 32));
  float sum = 0.f;
#pragma unroll
  for (int i = 0; i < 4; ++i) { vals[i] = __expf(vals[i] - mx); sum += vals[i]; }
#pragma unroll
  for (int o = 16; o > 0; o >>= 1) sum += __shfl_xor(sum, o, 32);
  float inv = 1.f / sum;
#pragma unroll
  for (int i = 0; i < 4; ++i) {
    int j = lane + i * 32;
    __bf16 a = (__bf16)(vals[i] * inv);
    if (j < 64) AH[(size_t)(((bb * 64 + w) * 64 + j)) * 64 + h] = a;
    else        AW[(size_t)(((bb * 64 + h) * 64 + (j - 64))) * 64 + w] = a;
  }
}

// ---- apply attention: ctx[c, pix] (+)= V(64c x 64k) * att(64k x 64n) -----
// 16 waves (4m x 4n), K looped 64 in two steps of 32. The attention tile is
// one contiguous 4KB block -> async copy; V rows are contiguous in the row-
// attention mode -> async copy there too, scalar gather otherwise.
__global__ void apply_att_k(const __bf16* __restrict__ V, const __bf16* __restrict__ A,
                            float* __restrict__ ctx, int fixedIsW, int accumulate) {
  __shared__ __align__(16) __bf16 Vs[64 * 32];    // m-major [mc][32]
  __shared__ __align__(16) __bf16 As2[32 * 64];   // k-major [k][64]
  int bb = blockIdx.x, fix = blockIdx.y, c_base = blockIdx.z * 64;
  int t = threadIdx.x, lane = t & 31, wid = t >> 5;
  int wm = wid & 3, wn = wid >> 2;
  int half = lane >> 4, ln = lane & 15;
  v8f acc = {};
  for (int k0 = 0; k0 < 64; k0 += 32) {
#if HAVE_ASYNC_LDS
    {  // att tile: contiguous 4KB -> flat b64 per lane (512 lanes * 8B)
      const char* g = (const char*)(A + (size_t)(((bb * 64 + fix) * 64 + k0)) * 64);
      char* l = (char*)As2;
      async_copy_b64(g + t * 8, l + t * 8);
    }
    if (!fixedIsW) {  // V rows contiguous: 64 rows x 64B -> b64 per lane
      int mc = t >> 3, chunk = t & 7;
      const char* g = (const char*)(V + (size_t)(bb * CDIM + c_base + mc) * NP +
                                    fix * WS + k0) + chunk * 8;
      char* l = (char*)Vs + mc * 64 + chunk * 8;
      async_copy_b64(g, l);
    } else {          // strided gather -> scalar staging
#pragma unroll
      for (int i = 0; i < 4; ++i) {
        int e = i * 512 + t, mc = e >> 5, k = e & 31;
        Vs[mc * 32 + k] = V[(size_t)(bb * CDIM + c_base + mc) * NP + (k0 + k) * WS + fix];
      }
    }
    wait_async0();
#else
#pragma unroll
    for (int i = 0; i < 4; ++i) {
      int e = i * 512 + t, mc = e >> 5, k = e & 31;
      int pix = fixedIsW ? ((k0 + k) * WS + fix) : (fix * WS + (k0 + k));
      Vs[mc * 32 + k] = V[(size_t)(bb * CDIM + c_base + mc) * NP + pix];
    }
#pragma unroll
    for (int i = 0; i < 4; ++i) {
      int e = i * 512 + t, k = e >> 6, n = e & 63;
      As2[k * 64 + n] = A[(size_t)(((bb * 64 + fix) * 64 + k0 + k)) * 64 + n];
    }
#endif
    __syncthreads();
    v16bf af  = load_frag_a(Vs, wm * 16 + ln, half);
    v16bf bf_ = load_frag_b_km(As2, wn * 16 + ln, half);
    acc = __builtin_amdgcn_wmma_f32_16x16x32_bf16(false, af, false, bf_, (short)0, acc, false, false);
    __syncthreads();
  }
  int n = wn * 16 + ln;
  int pix = fixedIsW ? (n * WS + fix) : (fix * WS + n);
#pragma unroll
  for (int r2 = 0; r2 < 8; ++r2) {
    int m = c_base + wm * 16 + half * 8 + r2;
    float* dst = &ctx[(size_t)(bb * CDIM + m) * NP + pix];
    if (accumulate) *dst += acc[r2]; else *dst = acc[r2];
  }
}

// ---- combine: (g1*cc1 + rot45(g2*cc2) + xb) * gate -> full-image bf16 ----
__global__ void combine_k(const float* __restrict__ ctx1, const float* __restrict__ ctx2,
                          const float* __restrict__ xb, const float* __restrict__ gca,
                          const float* __restrict__ g1p, const float* __restrict__ g2p,
                          __bf16* __restrict__ ctxb) {
  int idx = blockIdx.x * 256 + threadIdx.x;
  int bb = idx >> 20, r = idx & 0xFFFFF, c = r >> 12, p = r & 4095;
  int y = p >> 6, xx = p & 63;
  int b = bb >> 2, si = (bb >> 1) & 1, sj = bb & 1;
  float g1 = g1p[0], g2 = g2p[0];
  const float CT = 0.70710678118654752f;        // rotate(+45): ct=st=sqrt2/2
  float fx = (float)xx - 31.5f, fy = (float)y - 31.5f;
  float sx =  CT * fx + CT * fy + 31.5f;
  float sy = -CT * fx + CT * fy + 31.5f;
  float rx = rintf(sx), ry = rintf(sy);
  float o2 = 0.f;
  if (rx >= 0.f && rx < 64.f && ry >= 0.f && ry < 64.f)
    o2 = g2 * ctx2[(size_t)(bb * CDIM + c) * NP + (int)ry * WS + (int)rx];
  float val = g1 * ctx1[idx] + o2 + xb[idx];
  val *= gca[(b * CDIM + c) * 4 + si * 2 + sj];
  ctxb[((size_t)(b * CDIM + c) * HDIM + si * HS + y) * WDIM + sj * WS + xx] = (__bf16)val;
}

// ---- 3x3 conv (9-tap WMMA accumulation) + BN + residual + ReLU -----------
__global__ void conv3x3_k(const __bf16* __restrict__ ctxb, const float* __restrict__ cw,
                          const float* __restrict__ cb, const float* __restrict__ bw,
                          const float* __restrict__ bbias, const float* __restrict__ bmean,
                          const float* __restrict__ bvar, const float* __restrict__ gp,
                          const float* __restrict__ x, float* __restrict__ out) {
  __shared__ __bf16 As[32 * 32];
  __shared__ __bf16 Bs[64 * 32];
  int ng = blockIdx.x, mg = blockIdx.y, b = blockIdx.z;
  int yrow = ng >> 1, x0 = (ng & 1) * 64;
  int oc0 = mg * 32;
  int t = threadIdx.x, lane = t & 31, wid = t >> 5;
  int wm = wid & 1, wn = wid >> 1;
  int half = lane >> 4, ln = lane & 15;
  v8f acc = {};
  for (int tap = 0; tap < 9; ++tap) {
    int dy = tap / 3 - 1, dx = tap % 3 - 1;
    int yy = yrow + dy;
    for (int ks = 0; ks < 8; ++ks) {
      int c0 = ks * 32;
#pragma unroll
      for (int i = 0; i < 4; ++i) {
        int e = i * 256 + t, m = e >> 5, k = e & 31;
        As[m * 32 + k] = (__bf16)cw[((oc0 + m) * CDIM + c0 + k) * 9 + tap];
      }
#pragma unroll
      for (int i = 0; i < 8; ++i) {
        int e = i * 256 + t, n = e & 63, k = e >> 6;
        int xg = x0 + n + dx;
        __bf16 v = (__bf16)0.f;
        if (yy >= 0 && yy < HDIM && xg >= 0 && xg < WDIM)
          v = ctxb[((size_t)(b * CDIM + c0 + k) * HDIM + yy) * WDIM + xg];
        Bs[n * 32 + k] = v;
      }
      __syncthreads();
      v16bf af  = load_frag_a(As, wm * 16 + ln, half);
      v16bf bf_ = load_frag_b(Bs, wn * 16 + ln, half);
      acc = __builtin_amdgcn_wmma_f32_16x16x32_bf16(false, af, false, bf_, (short)0, acc, false, false);
      __syncthreads();
    }
  }
  float gamma = gp[0];
  int px = x0 + wn * 16 + ln;
#pragma unroll
  for (int r2 = 0; r2 < 8; ++r2) {
    int m = oc0 + wm * 16 + half * 8 + r2;
    float val = acc[r2] + cb[m];
    float sc = bw[m] * rsqrtf(bvar[m] + 1e-5f);
    val = (val - bmean[m]) * sc + bbias[m];
    size_t oi = ((size_t)(b * CDIM + m) * HDIM + yrow) * WDIM + px;
    out[oi] = fmaxf(gamma * val + x[oi], 0.f);
  }
}

// ---------------------------------------------------------------------------
extern "C" void kernel_launch(void* const* d_in, const int* in_sizes, int n_in,
                              void* d_out, int out_size, void* d_ws, size_t ws_size,
                              hipStream_t stream) {
  (void)in_sizes; (void)n_in; (void)out_size; (void)ws_size;
  const float* x   = (const float*)d_in[0];
  const float* wq  = (const float*)d_in[1];
  const float* bq  = (const float*)d_in[2];
  const float* wk  = (const float*)d_in[3];
  const float* bk  = (const float*)d_in[4];
  const float* wv  = (const float*)d_in[5];
  const float* bv  = (const float*)d_in[6];
  const float* g1  = (const float*)d_in[7];
  const float* g2  = (const float*)d_in[8];
  const float* nqw = (const float*)d_in[9];
  const float* nqb = (const float*)d_in[10];
  const float* nkw = (const float*)d_in[11];
  const float* nkb = (const float*)d_in[12];
  const float* nvw = (const float*)d_in[13];
  const float* nvb = (const float*)d_in[14];
  const float* nlg = (const float*)d_in[15];
  const float* cw  = (const float*)d_in[16];
  const float* cb  = (const float*)d_in[17];
  const float* bw  = (const float*)d_in[18];
  const float* bb_ = (const float*)d_in[19];
  const float* bm  = (const float*)d_in[20];
  const float* bvr = (const float*)d_in[21];
  const float* gm  = (const float*)d_in[22];

  // workspace layout (~0.79 GB total)
  char* ws = (char*)d_ws;
  size_t off = 0;
  auto alloc = [&](size_t bytes) -> char* {
    char* p = ws + off; off += (bytes + 255) & ~(size_t)255; return p;
  };
  float*  xb   = (float*)alloc(33554432ull * 4);   // blocks (32,256,64,64)
  float*  xr   = (float*)alloc(33554432ull * 4);   // rotated blocks
  __bf16* qb   = (__bf16*)alloc(4194304ull * 2);   // (32,32,4096)
  __bf16* kb   = (__bf16*)alloc(4194304ull * 2);
  __bf16* vb   = (__bf16*)alloc(33554432ull * 2);  // (32,256,4096)
  float*  eH   = (float*)alloc(8388608ull * 4);    // (32,64,64,64)
  float*  eW   = (float*)alloc(8388608ull * 4);
  __bf16* aH   = (__bf16*)alloc(8388608ull * 2);
  __bf16* aW   = (__bf16*)alloc(8388608ull * 2);
  float*  ctx1 = (float*)alloc(33554432ull * 4);
  float*  ctx2 = (float*)alloc(33554432ull * 4);
  __bf16* ctxb = (__bf16*)alloc(33554432ull * 2);  // full image (8,256,128,128)
  float*  pooled = (float*)alloc(8192ull * 4);
  float*  gca  = (float*)alloc(8192ull * 4);

  pool_max_k<<<8192, 256, 0, stream>>>(x, pooled);
  gca_k<<<BDIM, 256, 0, stream>>>(pooled, nqw, nqb, nkw, nkb, nvw, nvb, nlg, gca);
  extract_blocks_k<<<131072, 256, 0, stream>>>(x, xb);
  rotate_k<<<131072, 256, 0, stream>>>(xb, xr, 0.70710678118654752f, -0.70710678118654752f);

  for (int br = 0; br < 2; ++br) {
    const float* src = br ? xr : xb;
    float* ctx = br ? ctx2 : ctx1;
    gemm1x1_k<<<dim3(NBLK, 64, 1), 256, 0, stream>>>(wq, bq, src, qb, CR);
    gemm1x1_k<<<dim3(NBLK, 64, 1), 256, 0, stream>>>(wk, bk, src, kb, CR);
    gemm1x1_k<<<dim3(NBLK, 64, 8), 256, 0, stream>>>(wv, bv, src, vb, CDIM);
    energy_k<<<dim3(NBLK, 32), 256, 0, stream>>>(qb, kb, eH, 1);
    energy_k<<<dim3(NBLK, 32), 256, 0, stream>>>(qb, kb, eW, 0);
    softmax_k<<<16384, 256, 0, stream>>>(eH, eW, aH, aW);
    apply_att_k<<<dim3(NBLK, 64, 4), 512, 0, stream>>>(vb, aH, ctx, 1, 0);
    apply_att_k<<<dim3(NBLK, 64, 4), 512, 0, stream>>>(vb, aW, ctx, 0, 1);
  }

  combine_k<<<131072, 256, 0, stream>>>(ctx1, ctx2, xb, gca, g1, g2, ctxb);
  conv3x3_k<<<dim3(256, 8, 8), 256, 0, stream>>>(ctxb, cw, cb, bw, bb_, bm, bvr, gm,
                                                 x, (float*)d_out);
}
